// DynamicConv_58926951301261
// MI455X (gfx1250) — matrix operand order repaired and checked by reading, not compiled, not run on hardware
//
#include <hip/hip_runtime.h>

typedef __attribute__((ext_vector_type(16))) __bf16 bf16x16;
typedef __attribute__((ext_vector_type(8)))  float  f32x8;
typedef __attribute__((ext_vector_type(4)))  int    vint4;

#define B_      32
#define C_      128
#define H_      64
#define W_      64
#define OUT_    128
#define E_      4
#define INTERM_ 256
#define KDIM    1152   // C*3*3
#define WPS     147456 // OUT*C*3*3 weights per sample
#define XSTRIDE 136    // padded LDS row stride (elements) = 128 payload + 8 pad

#if __has_builtin(__builtin_amdgcn_global_load_async_to_lds_b128)
#define HAVE_ASYNC_LDS 1
#endif

// ---------------- Kernel 1: global average pool  x[B,C,H,W] -> pooled[B,C]
__global__ __launch_bounds__(256) void pool_kernel(const float* __restrict__ x,
                                                   float* __restrict__ pooled) {
  const int bc = blockIdx.x;                       // 0 .. B*C-1
  const float* p = x + (size_t)bc * (H_ * W_);
  float s = 0.f;
  for (int i = threadIdx.x; i < H_ * W_; i += 256) s += p[i];
  __shared__ float red[256];
  red[threadIdx.x] = s;
  __syncthreads();
  for (int off = 128; off > 0; off >>= 1) {
    if (threadIdx.x < off) red[threadIdx.x] += red[threadIdx.x + off];
    __syncthreads();
  }
  if (threadIdx.x == 0) pooled[bc] = red[0] * (1.0f / (H_ * W_));
}

// ---------------- Kernel 2: routing = avgpool( sigmoid(pooled @ rw^T + rb) )
__global__ __launch_bounds__(256) void routing_kernel(const float* __restrict__ pooled,
                                                      const float* __restrict__ rw,
                                                      const float* __restrict__ rb,
                                                      float* __restrict__ routing) {
  __shared__ float pl[B_ * C_];
  __shared__ float rt[B_][INTERM_];
  for (int i = threadIdx.x; i < B_ * C_; i += 256) pl[i] = pooled[i];
  __syncthreads();
  const int i = threadIdx.x;                       // interm index 0..255
  const float bias = rb[i];
  for (int b = 0; b < B_; ++b) {
    float acc = bias;
    for (int k = 0; k < C_; ++k) acc += pl[b * C_ + k] * rw[i * C_ + k];
    rt[b][i] = 1.0f / (1.0f + __expf(-acc));
  }
  __syncthreads();
  if (threadIdx.x < B_ * E_) {
    const int b = threadIdx.x >> 2, e = threadIdx.x & 3;
    float s = 0.f;
    for (int q = 0; q < INTERM_ / E_; ++q) s += rt[b][e * (INTERM_ / E_) + q];
    routing[b * E_ + e] = s * (float)E_ / (float)INTERM_;
  }
}

// ---------------- Kernel 3: per-sample mixed kernels (bf16), permuted K-order
// output layout: wmix[b][m][k'] with k' = p*128 + c, p = kh*3+kw
// source flat j = m*1152 + c*9 + p  (original OIHW flatten)
__global__ __launch_bounds__(256) void mix_kernel(const float* __restrict__ ew,
                                                  const float* __restrict__ routing,
                                                  __bf16* __restrict__ wmix) {
  const int b = blockIdx.y;
  const int g = blockIdx.x * 256 + threadIdx.x;    // 0 .. 18431
  const int c0 = (g & 15) * 8;                     // 8 consecutive channels
  const int p  = (g >> 4) % 9;
  const int m  = (g >> 4) / 9;
  const float r0 = routing[b * E_ + 0];
  const float r1 = routing[b * E_ + 1];
  const float r2 = routing[b * E_ + 2];
  const float r3 = routing[b * E_ + 3];
  union { __bf16 h[8]; uint4 u; } o;
#pragma unroll
  for (int i = 0; i < 8; ++i) {
    const size_t j = (size_t)m * 1152 + (size_t)(c0 + i) * 9 + p;
    const float s = r0 * ew[j] + r1 * ew[(size_t)WPS + j] +
                    r2 * ew[2 * (size_t)WPS + j] + r3 * ew[3 * (size_t)WPS + j];
    o.h[i] = (__bf16)s;
  }
  *(uint4*)(wmix + (size_t)b * WPS + (size_t)m * KDIM + p * 128 + c0) = o.u;
}

// ---------------- Kernel 4: x (f32, NCHW) -> xbn (bf16, NHWC)  grid=(H,B)
__global__ __launch_bounds__(256) void cvt_nhwc_kernel(const float* __restrict__ x,
                                                       __bf16* __restrict__ xbn) {
  __shared__ __bf16 T[W_ * C_];                    // [w][c], 16 KB
  const int b = blockIdx.y, h = blockIdx.x;
  const int c  = threadIdx.x >> 1;                 // 0..127
  const int w0 = (threadIdx.x & 1) * 32;
  const float* src = x + (((size_t)(b * C_ + c) * H_) + h) * W_ + w0;
#pragma unroll
  for (int q = 0; q < 8; ++q) {
    const float4 v = *(const float4*)(src + q * 4);
    T[(w0 + q * 4 + 0) * C_ + c] = (__bf16)v.x;
    T[(w0 + q * 4 + 1) * C_ + c] = (__bf16)v.y;
    T[(w0 + q * 4 + 2) * C_ + c] = (__bf16)v.z;
    T[(w0 + q * 4 + 3) * C_ + c] = (__bf16)v.w;
  }
  __syncthreads();
  __bf16* dst = xbn + ((size_t)(b * H_ + h)) * (size_t)(W_ * C_);
#pragma unroll
  for (int q = 0; q < 4; ++q) {
    const int e = (threadIdx.x * 4 + q) * 8;       // element index, 8 bf16 per uint4
    *(uint4*)(dst + e) = *(const uint4*)(T + e);
  }
}

// ---------------- Kernel 5: implicit-GEMM CondConv via WMMA bf16
// grid = (H, B); block = 256 (8 waves). Block computes Y[b, 0..127, h, 0..63].
// K-order: k' = (kh*3+kw)*128 + c  -> B fragments read straight from LDS.
__global__ __launch_bounds__(256) void condconv_kernel(const __bf16* __restrict__ xbn,
                                                       const __bf16* __restrict__ wmix,
                                                       float* __restrict__ out) {
  // XsT[kh][wi][c] : wi = w'+1 for w' in [-1,64] (66 slots), c padded to 136
  __shared__ __bf16 XsT[3 * 66 * XSTRIDE];         // 53,856 B

  const int tid = threadIdx.x;
  const int b = blockIdx.y;
  const int h = blockIdx.x;

  // ---- stage 3 channel-major input rows, halo/OOB -> 0
  for (int idx = tid; idx < 3 * 66 * 17; idx += 256) {
    const int c8 = idx % 17;                       // 17th chunk = pad lane
    const int rw = idx / 17;                       // kh*66 + wi
    const int kh = rw / 66, wi = rw - kh * 66;
    const int hy = h - 1 + kh;
    const bool inr = (c8 < 16) && (wi >= 1) && (wi <= 64) && (hy >= 0) && (hy < H_);
#ifdef HAVE_ASYNC_LDS
    if (inr) {
      __builtin_amdgcn_global_load_async_to_lds_b128(
          (__attribute__((address_space(1))) vint4*)
              (xbn + (((size_t)(b * H_ + hy) * W_) + (wi - 1)) * C_ + c8 * 8),
          (__attribute__((address_space(3))) vint4*)&XsT[rw * XSTRIDE + c8 * 8],
          0, 0);
    } else {
      const uint4 z = {0u, 0u, 0u, 0u};
      *(uint4*)&XsT[rw * XSTRIDE + c8 * 8] = z;
    }
#else
    uint4 v = {0u, 0u, 0u, 0u};
    if (inr)
      v = *(const uint4*)(xbn + (((size_t)(b * H_ + hy) * W_) + (wi - 1)) * C_ + c8 * 8);
    *(uint4*)&XsT[rw * XSTRIDE + c8 * 8] = v;
#endif
  }
#ifdef HAVE_ASYNC_LDS
#if __has_builtin(__builtin_amdgcn_s_wait_asynccnt)
  __builtin_amdgcn_s_wait_asynccnt(0);
#else
  asm volatile("s_wait_asynccnt 0x0" ::: "memory");
#endif
#endif
  __syncthreads();

  const int wave = tid >> 5;
  const int lane = tid & 31;
  const int lane16 = lane & 15;
  const int half = lane >> 4;

  // A fragments (ISA 16-bit A map): elems 0..7 -> k = 8*half+0..7, elems 8..15 -> +16
  const __bf16* wbase = wmix + (size_t)b * WPS +
                        (size_t)(wave * 16 + lane16) * KDIM + half * 8;
  // B fragments: same K order as A: elems 0..7 -> k = 8*half+0..7, elems 8..15 -> +16
  const int laneoffB = lane16 * XSTRIDE + half * 8;

  f32x8 acc[4] = {f32x8{0.f,0.f,0.f,0.f,0.f,0.f,0.f,0.f},
                  f32x8{0.f,0.f,0.f,0.f,0.f,0.f,0.f,0.f},
                  f32x8{0.f,0.f,0.f,0.f,0.f,0.f,0.f,0.f},
                  f32x8{0.f,0.f,0.f,0.f,0.f,0.f,0.f,0.f}};

#pragma unroll 1
  for (int p = 0; p < 9; ++p) {                    // kernel position: p = kh*3+kw
    const int kh = p / 3, kw = p - kh * 3;
    const __bf16* xrow = &XsT[(kh * 66 + kw) * XSTRIDE + laneoffB];
    if (p < 8) __builtin_prefetch(wbase + (p + 1) * 128, 0, 0);

#pragma unroll
    for (int cs = 0; cs < 4; ++cs) {               // 32-channel K-slices
      const int kp = p * 128 + cs * 32;

      bf16x16 afrag;
      {
        union { uint4 u[2]; bf16x16 v; } ua;
        ua.u[0] = *(const uint4*)(wbase + kp);
        ua.u[1] = *(const uint4*)(wbase + kp + 16);
        afrag = ua.v;
      }

      bf16x16 bfrag[4];
#pragma unroll
      for (int j = 0; j < 4; ++j) {                // N-tiles of 16 columns
        union { uint4 u[2]; bf16x16 v; } ub;
        const __bf16* bp = xrow + j * 16 * XSTRIDE + cs * 32;
        ub.u[0] = *(const uint4*)bp;               // K = 8*half + 0..7
        ub.u[1] = *(const uint4*)(bp + 16);        // K = 16 + 8*half + 0..7
        bfrag[j] = ub.v;
      }
#pragma unroll
      for (int j = 0; j < 4; ++j) {
        acc[j] = __builtin_amdgcn_wmma_f32_16x16x32_bf16(
            /*neg_a=*/false, afrag, /*neg_b=*/false, bfrag[j],
            /*c_mod=*/(short)0, acc[j], /*reuse_a=*/false, /*reuse_b=*/false);
      }
    }
  }

  // ---- epilogue: D layout -> out[b, m, h, n]
#pragma unroll
  for (int j = 0; j < 4; ++j) {
#pragma unroll
    for (int e = 0; e < 8; ++e) {
      const int m = wave * 16 + e + 8 * half;
      const int n = j * 16 + lane16;
      out[(((size_t)b * OUT_ + m) * H_ + h) * W_ + n] = acc[j][e];
    }
  }
}

extern "C" void kernel_launch(void* const* d_in, const int* in_sizes, int n_in,
                              void* d_out, int out_size, void* d_ws, size_t ws_size,
                              hipStream_t stream) {
  const float* x             = (const float*)d_in[0];
  const float* routing_w     = (const float*)d_in[1];
  const float* routing_b     = (const float*)d_in[2];
  const float* expert_weight = (const float*)d_in[3];
  float* out = (float*)d_out;

  char* ws = (char*)d_ws;
  __bf16* xbn    = (__bf16*)ws;                               // 33,554,432 B (NHWC bf16)
  __bf16* wmix   = (__bf16*)(ws + 33554432);                  //  9,437,184 B
  float*  pooled = (float*)(ws + 33554432 + 9437184);         //     16,384 B
  float*  routing= (float*)(ws + 33554432 + 9437184 + 16384); //        512 B

  pool_kernel<<<dim3(B_ * C_), 256, 0, stream>>>(x, pooled);
  routing_kernel<<<dim3(1), 256, 0, stream>>>(pooled, routing_w, routing_b, routing);
  mix_kernel<<<dim3(72, B_), 256, 0, stream>>>(expert_weight, routing, wmix);
  cvt_nhwc_kernel<<<dim3(H_, B_), 256, 0, stream>>>(x, xbn);
  condconv_kernel<<<dim3(H_, B_), 256, 0, stream>>>(xbn, wmix, out);
}
